// Attention_40140764348780
// MI455X (gfx1250) — compile-verified, matrix operand order checked by reading, not run on hardware
//
#include <hip/hip_runtime.h>
#include <math.h>

// ---------------------------------------------------------------------------
// Relative-position attention for MI455X (gfx1250), wave32 + WMMA bf16.
//   B=4, L=1024, D=128, DV=192, MAXREL=64 (129 relative buckets)
// ---------------------------------------------------------------------------

typedef __bf16 bf16_t;
typedef __attribute__((ext_vector_type(16))) __bf16 v16bf;
typedef __attribute__((ext_vector_type(8)))  float  v8f;

#define B_       4
#define L_       1024
#define D_       128
#define DV_      192
#define NR_      129          // 2*MAXREL+1 bucket count
#define MAXREL_  64
#define BLOCK_M  64           // q rows per workgroup (4 waves x 16)
#define KB       32           // attention-k columns per main-loop iteration
#define WAVES    4
#define RSTRIDE  132          // padded row stride for R table in LDS (floats)
#define SBUCK    160          // bucket count padded to multiple of 32

// ---- dynamic LDS layout (bytes) -------------------------------------------
#define OFF_KS   0
#define SZ_KS    (KB * D_ * 2)                 // K block, bf16: 8 KB
#define OFF_VS   (OFF_KS + SZ_KS)
#define SZ_VS    (KB * DV_ * 2)                // V block, bf16: 12 KB
#define OFF_R    (OFF_VS + SZ_VS)
#define SZ_R     (WAVES * 16 * RSTRIDE * 4)    // R[q][bucket] f32: 33 KB
#define OFF_S    (OFF_R + SZ_R)
#define SZ_S     (WAVES * 16 * SBUCK * 4)      // bucket sums f32: 40 KB
#define OFF_P    (OFF_S + SZ_S)
#define SZ_P     (WAVES * 16 * KB * 2)         // P tile bf16: 4 KB
#define LDS_TOTAL (OFF_P + SZ_P)               // 99328 bytes

// Per-lane element -> K-index mapping shared by the 16-bit A and B WMMA
// layouts (ISA 7.12.2): element e of the v16 register holds K = kmap(e, half)
// where half = (lane >= 16).
__device__ __forceinline__ int kmap(int e, int hf) {
  return (e & 7) + ((e >> 3) << 4) + (hf << 3);
}

__device__ __forceinline__ v8f wmma_bf16(v16bf a, v16bf b, v8f c) {
  // emits v_wmma_f32_16x16x32_bf16
  return __builtin_amdgcn_wmma_f32_16x16x32_bf16(
      /*neg_a=*/false, a, /*neg_b=*/false, b,
      /*c_mod=*/(short)0, c, /*reuse_a=*/false, /*reuse_b=*/false);
}

extern "C" __global__ __launch_bounds__(128)
void relattn_kernel(const float* __restrict__ Q,
                    const float* __restrict__ V,
                    const float* __restrict__ Kg,
                    const float* __restrict__ relK,
                    const float* __restrict__ relV,
                    float* __restrict__ out) {
  extern __shared__ char smem[];
  bf16_t* Ks = (bf16_t*)(smem + OFF_KS);   // [KB][D_]
  bf16_t* Vs = (bf16_t*)(smem + OFF_VS);   // [KB][DV_]
  float*  Rl = (float*) (smem + OFF_R);    // [WAVES][16][RSTRIDE]
  float*  Sl = (float*) (smem + OFF_S);    // [WAVES][16][SBUCK]
  bf16_t* Pl = (bf16_t*)(smem + OFF_P);    // [WAVES][16][KB]

  const int tid  = threadIdx.x;
  const int lane = tid & 31;
  const int wv   = tid >> 5;
  const int hf   = lane >> 4;       // which 16-lane half
  const int ln16 = lane & 15;
  const int b    = blockIdx.x;
  const int q0   = blockIdx.y * BLOCK_M;
  const int qrow = q0 + wv * 16;    // first q row owned by this wave

  // ---- Q tile (16 x 128) into WMMA A-layout registers, bf16 ----------------
  v16bf qa[4];
  {
    const float* qb = Q + ((size_t)b * L_ + qrow + ln16) * D_;
#pragma unroll
    for (int c = 0; c < 4; ++c)
#pragma unroll
      for (int e = 0; e < 16; ++e)
        qa[c][e] = (bf16_t)qb[c * 32 + kmap(e, hf)];
  }

  // ---- R[q_local][r] = Q . rel_keys_table[r]  (16x128 @ 128x129, WMMA) -----
  {
    float* Rw = Rl + (size_t)wv * 16 * RSTRIDE;
    const int rcol = ln16;  // column within tile
#pragma unroll
    for (int rt = 0; rt < 9; ++rt) {
      v8f acc = {};
      const int r = rt * 16 + rcol;
#pragma unroll
      for (int c = 0; c < 4; ++c) {
        v16bf bm;
#pragma unroll
        for (int e = 0; e < 16; ++e) {
          bf16_t val = (bf16_t)0.0f;
          if (r < NR_) val = (bf16_t)relK[(size_t)r * D_ + c * 32 + kmap(e, hf)];
          bm[e] = val;
        }
        acc = wmma_bf16(qa[c], bm, acc);
      }
      if (r < NR_) {
#pragma unroll
        for (int v = 0; v < 8; ++v)
          Rw[(v + 8 * hf) * RSTRIDE + r] = acc[v];
      }
    }
  }

  // ---- Pass 1: row max / sum (flash-style, per-lane partials) --------------
  float mrow[8], lrow[8];
#pragma unroll
  for (int v = 0; v < 8; ++v) { mrow[v] = -1e30f; lrow[v] = 0.0f; }

  for (int kb0 = 0; kb0 < L_; kb0 += KB) {
    __syncthreads();
    for (int i = tid; i < KB * D_; i += 128) {
      int r = i >> 7, cd = i & (D_ - 1);
      Ks[i] = (bf16_t)Kg[((size_t)b * L_ + kb0 + r) * D_ + cd];
    }
    __syncthreads();

#pragma unroll
    for (int kt = 0; kt < 2; ++kt) {
      v8f s = {};
#pragma unroll
      for (int c = 0; c < 4; ++c) {
        v16bf bm;
#pragma unroll
        for (int e = 0; e < 16; ++e)
          bm[e] = Ks[(kt * 16 + ln16) * D_ + c * 32 + kmap(e, hf)];
        s = wmma_bf16(qa[c], bm, s);
      }
#pragma unroll
      for (int v = 0; v < 8; ++v) {
        const int row  = v + 8 * hf;
        const int q    = qrow + row;
        const int k    = kb0 + kt * 16 + ln16;
        const int dq   = k - q;
        const int dqc  = dq < -MAXREL_ ? -MAXREL_ : (dq > MAXREL_ ? MAXREL_ : dq);
        const int ridx = dqc + MAXREL_;
        const float ad = (float)(dq < 0 ? -dq : dq);
        float x = s[v] + Rl[((size_t)wv * 16 + row) * RSTRIDE + ridx]
                  - __logf(1.0f + ad);
        float mn = fmaxf(mrow[v], x);
        lrow[v]  = lrow[v] * __expf(mrow[v] - mn) + __expf(x - mn);
        mrow[v]  = mn;
      }
    }
  }

  // merge (m,l) across the 16-lane group (masks 1..8 keep halves separate)
#pragma unroll
  for (int v = 0; v < 8; ++v) {
    float m = mrow[v], l = lrow[v];
#pragma unroll
    for (int off = 1; off < 16; off <<= 1) {
      float mo = __shfl_xor(m, off, 32);
      float lo = __shfl_xor(l, off, 32);
      float mn = fmaxf(m, mo);
      l = l * __expf(m - mn) + lo * __expf(mo - mn);
      m = mn;
    }
    mrow[v] = m;
    lrow[v] = 1.0f / l;   // store inverse sum
  }

  // ---- Pass 2: probs, Z = P@V, bucket sums, Z += s@relV --------------------
  float*  Sw = Sl + (size_t)wv * 16 * SBUCK;
  bf16_t* Pw = Pl + (size_t)wv * 16 * KB;
  for (int i = lane; i < 16 * SBUCK; i += 32) Sw[i] = 0.0f;   // per-wave region

  v8f zacc[12] = {};

  for (int kb0 = 0; kb0 < L_; kb0 += KB) {
    __syncthreads();
    for (int i = tid; i < KB * D_; i += 128) {
      int r = i >> 7, cd = i & (D_ - 1);
      Ks[i] = (bf16_t)Kg[((size_t)b * L_ + kb0 + r) * D_ + cd];
    }
    for (int r = 0; r < KB; ++r)
      for (int j = tid; j < DV_; j += 128)
        Vs[r * DV_ + j] = (bf16_t)V[((size_t)b * L_ + kb0 + r) * DV_ + j];
    __syncthreads();

#pragma unroll
    for (int kt = 0; kt < 2; ++kt) {
      v8f s = {};
#pragma unroll
      for (int c = 0; c < 4; ++c) {
        v16bf bm;
#pragma unroll
        for (int e = 0; e < 16; ++e)
          bm[e] = Ks[(kt * 16 + ln16) * D_ + c * 32 + kmap(e, hf)];
        s = wmma_bf16(qa[c], bm, s);
      }
#pragma unroll
      for (int v = 0; v < 8; ++v) {
        const int row  = v + 8 * hf;
        const int q    = qrow + row;
        const int k    = kb0 + kt * 16 + ln16;
        const int dq   = k - q;
        const int dqc  = dq < -MAXREL_ ? -MAXREL_ : (dq > MAXREL_ ? MAXREL_ : dq);
        const int ridx = dqc + MAXREL_;
        const float ad = (float)(dq < 0 ? -dq : dq);
        float x = s[v] + Rl[((size_t)wv * 16 + row) * RSTRIDE + ridx]
                  - __logf(1.0f + ad);
        float p = __expf(x - mrow[v]) * lrow[v];       // final normalized prob
        atomicAdd(&Sw[row * SBUCK + ridx], p);          // ds_add_f32 bucket sum
        Pw[row * KB + kt * 16 + ln16] = (bf16_t)p;      // C-layout -> LDS
      }
    }

    // reload P in A-layout (same-wave LDS round trip) and do Z += P @ V
    v16bf pa;
#pragma unroll
    for (int e = 0; e < 16; ++e)
      pa[e] = Pw[ln16 * KB + kmap(e, hf)];
#pragma unroll
    for (int nt = 0; nt < 12; ++nt) {
      v16bf bm;
#pragma unroll
      for (int e = 0; e < 16; ++e)
        bm[e] = Vs[kmap(e, hf) * DV_ + nt * 16 + ln16];
      zacc[nt] = wmma_bf16(pa, bm, zacc[nt]);
    }
  }

  // ---- Z += s @ rel_values_table  (16 x 160 @ 160 x 192, pad rows = 0) -----
#pragma unroll
  for (int c = 0; c < SBUCK / 32; ++c) {
    v16bf pa;
#pragma unroll
    for (int e = 0; e < 16; ++e)
      pa[e] = (bf16_t)Sw[ln16 * SBUCK + c * 32 + kmap(e, hf)];
#pragma unroll
    for (int nt = 0; nt < 12; ++nt) {
      v16bf bm;
#pragma unroll
      for (int e = 0; e < 16; ++e) {
        const int r = c * 32 + kmap(e, hf);
        bf16_t val = (bf16_t)0.0f;
        if (r < NR_) val = (bf16_t)relV[(size_t)r * DV_ + nt * 16 + ln16];
        bm[e] = val;
      }
      zacc[nt] = wmma_bf16(pa, bm, zacc[nt]);
    }
  }

  // ---- write Z (C-layout -> global) ----------------------------------------
#pragma unroll
  for (int nt = 0; nt < 12; ++nt)
#pragma unroll
    for (int v = 0; v < 8; ++v) {
      const int row = v + 8 * hf;
      out[((size_t)b * L_ + qrow + row) * DV_ + nt * 16 + ln16] = zacc[nt][v];
    }
}

extern "C" void kernel_launch(void* const* d_in, const int* in_sizes, int n_in,
                              void* d_out, int out_size, void* d_ws, size_t ws_size,
                              hipStream_t stream) {
  (void)in_sizes; (void)n_in; (void)out_size; (void)d_ws; (void)ws_size;
  const float* Q    = (const float*)d_in[0];  // query            [B,L,D]
  const float* V    = (const float*)d_in[1];  // value            [B,L,DV]
  const float* K    = (const float*)d_in[2];  // key_in           [B,L,D]
  const float* relK = (const float*)d_in[3];  // rel_keys_table   [129,D]
  const float* relV = (const float*)d_in[4];  // rel_values_table [129,DV]
  float* out = (float*)d_out;                 // z                [B,L,DV]

  dim3 grid(B_, L_ / BLOCK_M);  // (4, 16)
  dim3 block(128);              // 4 waves of 32
  relattn_kernel<<<grid, block, LDS_TOTAL, stream>>>(Q, V, K, relK, relV, out);
}
